// RPEAttention_62277025792790
// MI455X (gfx1250) — compile-verified, matrix operand order checked by reading, not run on hardware
//
#include <hip/hip_runtime.h>

typedef __attribute__((ext_vector_type(16))) __bf16 v16bf;
typedef __attribute__((ext_vector_type(8)))  __bf16 v8bf;
typedef __attribute__((ext_vector_type(8)))  float  v8f;

#define DEV __device__ __forceinline__

// Problem dims
constexpr int B_   = 4;
constexpr int Cc   = 384;
constexpr int HH   = 8;
constexpr int HD   = 48;
constexpr int NN   = 1024;
constexpr int C3   = 1152;
constexpr int LDH  = 64;    // head dim padded for K-dim of WMMA
constexpr int LPAD = 240;   // bucket count padded for N tiles
constexpr int LK   = 256;   // bucket count padded for K tiles
constexpr int LBK  = 225;
constexpr float SCALE = 0.14433756729740643f;  // 48^-0.5

DEV int clamp7(int v) { return v < -7 ? -7 : (v > 7 ? 7 : v); }

// Load one 16x32 bf16 A/B fragment from K-contiguous [16][ld] storage.
// Lanes 0-15: row=lane, K = 0..7,16..23 ; lanes 16-31: row=lane-16, K = 8..15,24..31.
DEV v16bf load_frag(const __bf16* base, int ld, int lane) {
  const __bf16* p = base + (long)(lane & 15) * ld + ((lane >> 4) << 3);
  v8bf lo = *(const v8bf*)(p);
  v8bf hi = *(const v8bf*)(p + 16);
  return __builtin_shufflevector(lo, hi, 0,1,2,3,4,5,6,7,8,9,10,11,12,13,14,15);
}

DEV v8f wmma_bf16(v16bf a, v16bf b, v8f c) {
  return __builtin_amdgcn_wmma_f32_16x16x32_bf16(false, a, false, b, (short)0, c, false, false);
}

// ---------------- workspace zero ----------------
__global__ void k_zero(uint4* p, long n16) {
  long i = (long)blockIdx.x * blockDim.x + threadIdx.x;
  long stride = (long)gridDim.x * blockDim.x;
  uint4 z{0, 0, 0, 0};
  for (; i < n16; i += stride) p[i] = z;
}

// ---------------- conversions ----------------
__global__ void k_cvt_wqkv(const float* w, __bf16* WT) {
  int i = blockIdx.x * blockDim.x + threadIdx.x;
  if (i >= C3 * Cc) return;
  int o = i / Cc, c = i % Cc;
  WT[i] = (__bf16)w[c * C3 + o];
}
__global__ void k_cvt_wproj(const float* w, __bf16* WT) {
  int i = blockIdx.x * blockDim.x + threadIdx.x;
  if (i >= Cc * Cc) return;
  int o = i / Cc, c = i % Cc;
  WT[i] = (__bf16)w[c * Cc + o];
}
__global__ void k_cvt_tab(const float* tq, const float* tk, const float* tv,
                          __bf16* TQ, __bf16* TK, __bf16* TVT) {
  int i = blockIdx.x * blockDim.x + threadIdx.x;
  if (i >= HH * LBK * HD) return;
  int d = i % HD; int t = i / HD; int Lb = t % LBK; int h = t / LBK;
  TQ[(h * LPAD + Lb) * LDH + d] = (__bf16)tq[i];
  TK[(h * LPAD + Lb) * LDH + d] = (__bf16)tk[i];
  TVT[((long)(h * HD + d)) * LK + Lb] = (__bf16)tv[i];
}
__global__ void k_cvt_x(const float* x, __bf16* XT) {
  int i = blockIdx.x * blockDim.x + threadIdx.x;
  if (i >= B_ * NN * Cc) return;
  int c = i % Cc; int t = i / Cc; int n = t % NN; int b = t / NN;
  XT[i] = (__bf16)x[((long)(b * Cc + c)) * NN + n];
}

// ---------------- QKV projection GEMM: [4096,384] x [384,1152] ----------------
__global__ void __launch_bounds__(256) k_qkv(const __bf16* XT, const __bf16* WQ,
                                             __bf16* Q, __bf16* K, __bf16* VT) {
  int lane = threadIdx.x & 31, w = threadIdx.x >> 5;
  int m0 = blockIdx.x * 16;
  int n0 = blockIdx.y * 128 + w * 16;
  v8f acc = {};
  for (int k0 = 0; k0 < Cc; k0 += 32) {
    v16bf a = load_frag(XT + (long)m0 * Cc + k0, Cc, lane);
    v16bf b = load_frag(WQ + (long)n0 * Cc + k0, Cc, lane);
    acc = wmma_bf16(a, b, acc);
  }
  int col = lane & 15, rb = (lane >> 4) << 3;
  int o = n0 + col;
  int t = o / Cc, cc = o % Cc, h = cc / HD, d = cc % HD;
  for (int r = 0; r < 8; ++r) {
    int gi = m0 + rb + r;
    int b_ = gi >> 10, ii = gi & (NN - 1);
    float v = acc[r];
    long base = ((long)(b_ * HH + h) * NN + ii) * LDH + d;
    if (t == 0)      Q[base] = (__bf16)(v * SCALE);
    else if (t == 1) K[base] = (__bf16)v;
    else             VT[((long)(b_ * HH + h) * HD + d) * NN + ii] = (__bf16)v;
  }
}

// ---------------- RPE GEMMs: sk = q@tk^T, sq = scale*(k@tq^T) ----------------
__global__ void __launch_bounds__(256) k_rpe(const __bf16* Q, const __bf16* K,
                                             const __bf16* TK, const __bf16* TQ,
                                             __bf16* SK, __bf16* SQ) {
  int lane = threadIdx.x & 31, w = threadIdx.x >> 5;
  int task = blockIdx.x * 8 + w;              // 32*64*15 tasks
  int nt = task % 15; int mt = (task / 15) & 63; int bh = task / (15 * 64);
  int h = bh & 7;
  const __bf16* aq = Q + ((long)bh * NN + mt * 16) * LDH;
  const __bf16* ak = K + ((long)bh * NN + mt * 16) * LDH;
  const __bf16* bk = TK + (long)(h * LPAD + nt * 16) * LDH;
  const __bf16* bq = TQ + (long)(h * LPAD + nt * 16) * LDH;
  v8f sk = {}, sq = {};
  for (int k0 = 0; k0 < LDH; k0 += 32) {
    v16bf fa = load_frag(aq + k0, LDH, lane);
    v16bf fk = load_frag(ak + k0, LDH, lane);
    v16bf fb = load_frag(bk + k0, LDH, lane);
    v16bf fq = load_frag(bq + k0, LDH, lane);
    sk = wmma_bf16(fa, fb, sk);
    sq = wmma_bf16(fk, fq, sq);
  }
  int col = lane & 15, rb = (lane >> 4) << 3;
  for (int r = 0; r < 8; ++r) {
    int i = mt * 16 + rb + r, n = nt * 16 + col;
    SK[((long)bh * NN + i) * LPAD + n] = (__bf16)sk[r];
    SQ[((long)bh * NN + i) * LPAD + n] = (__bf16)(sq[r] * SCALE);
  }
}

// ---------------- pass 1: softmax row statistics (m, l) ----------------
__global__ void __launch_bounds__(256) k_pass1(const __bf16* Q, const __bf16* K,
                                               const __bf16* SK, const __bf16* SQ,
                                               float* M, float* Lr) {
  __shared__ __bf16 skT[8][16 * LPAD];        // per-wave SK slice (i-rows x buckets)
  int lane = threadIdx.x & 31, w = threadIdx.x >> 5;
  int task = blockIdx.x * 8 + w;              // 2048 tasks
  int it = task & 63, bh = task >> 6;
  int i0 = it * 16;
  // stage SK[i0..i0+15][0..239] into LDS once
  for (int idx = lane; idx < 16 * LPAD; idx += 32) {
    int rr = idx / LPAD, cc2 = idx - rr * LPAD;
    skT[w][idx] = SK[((long)bh * NN + i0 + rr) * LPAD + cc2];
  }
  const __bf16* qb = Q + ((long)bh * NN + i0) * LDH;
  v16bf a0 = load_frag(qb, LDH, lane);
  v16bf a1 = load_frag(qb + 32, LDH, lane);
  int col = lane & 15, rb = (lane >> 4) << 3;
  float runm[8], runs[8];
  for (int r = 0; r < 8; ++r) { runm[r] = -1e30f; runs[r] = 0.f; }
  __syncthreads();
  for (int jt = 0; jt < 64; ++jt) {
    int j0 = jt * 16;
    const __bf16* kb = K + ((long)bh * NN + j0) * LDH;
    v16bf b0 = load_frag(kb, LDH, lane);
    v16bf b1 = load_frag(kb + 32, LDH, lane);
    v8f s = {};
    s = wmma_bf16(a0, b0, s);
    s = wmma_bf16(a1, b1, s);
    int j = j0 + col, yj = j >> 5, xj = j & 31;
    for (int r = 0; r < 8; ++r) {
      int i = i0 + rb + r, yi = i >> 5, xi = i & 31;
      int dy = clamp7(yi - yj), dx = clamp7(xi - xj);
      int bk1 = (dy + 7) * 15 + (dx + 7);
      int bk2 = 224 - bk1;                    // (7-dy)*15 + (7-dx)
      float v = s[r]
          + (float)skT[w][(rb + r) * LPAD + bk1]
          + (float)SQ[((long)bh * NN + j) * LPAD + bk2];
      float tm = v;
      for (int m = 1; m < 16; m <<= 1) tm = fmaxf(tm, __shfl_xor(tm, m, 32));
      float nm = fmaxf(runm[r], tm);
      float e = __expf(v - nm);
      for (int m = 1; m < 16; m <<= 1) e += __shfl_xor(e, m, 32);
      runs[r] = runs[r] * __expf(runm[r] - nm) + e;
      runm[r] = nm;
    }
  }
  for (int r = 0; r < 8; ++r) {
    if (col == r) {
      int i = i0 + rb + r;
      M[(long)bh * NN + i]  = runm[r];
      Lr[(long)bh * NN + i] = runs[r];
    }
  }
}

// ---------------- pass 2: P = softmax(S); out = P@V + wsum@table_v ----------------
__global__ void __launch_bounds__(32) k_pass2(const __bf16* Q, const __bf16* K,
                                              const __bf16* VT, const __bf16* SK,
                                              const __bf16* SQ, const __bf16* TVT,
                                              const float* M, const float* Lr,
                                              __bf16* OUTP) {
  __shared__ float  wsF[16 * 256];            // fp32 bucket accumulators
  __shared__ __bf16 wsB[16 * 256];            // bf16 copy for WMMA A operand
  __shared__ __bf16 pst[16 * 32];             // P tile relayout C-frag -> A-frag
  __shared__ __bf16 skT[16 * LPAD];           // staged SK slice
  int lane = threadIdx.x & 31;
  int task = blockIdx.x;                      // 2048 tasks, 1 wave each
  int it = task & 63, bh = task >> 6;
  int h = bh & 7, b_ = bh >> 3;
  int i0 = it * 16;
  for (int idx = lane; idx < 16 * 256; idx += 32) wsF[idx] = 0.f;
  for (int idx = lane; idx < 16 * LPAD; idx += 32) {
    int rr = idx / LPAD, cc2 = idx - rr * LPAD;
    skT[idx] = SK[((long)bh * NN + i0 + rr) * LPAD + cc2];
  }
  const __bf16* qb = Q + ((long)bh * NN + i0) * LDH;
  v16bf a0 = load_frag(qb, LDH, lane);
  v16bf a1 = load_frag(qb + 32, LDH, lane);
  int col = lane & 15, rb = (lane >> 4) << 3;
  float rowm[8], rinv[8];
  for (int r = 0; r < 8; ++r) {
    int i = i0 + rb + r;
    rowm[r] = M[(long)bh * NN + i];
    rinv[r] = 1.0f / Lr[(long)bh * NN + i];
  }
  v8f oacc[3] = {{}, {}, {}};
  __syncthreads();
  for (int jt = 0; jt < 32; ++jt) {
    for (int sub = 0; sub < 2; ++sub) {
      int j0 = jt * 32 + sub * 16;
      const __bf16* kb = K + ((long)bh * NN + j0) * LDH;
      v16bf b0 = load_frag(kb, LDH, lane);
      v16bf b1 = load_frag(kb + 32, LDH, lane);
      v8f s = {};
      s = wmma_bf16(a0, b0, s);
      s = wmma_bf16(a1, b1, s);
      int j = j0 + col, yj = j >> 5, xj = j & 31;
      for (int r = 0; r < 8; ++r) {
        int i = i0 + rb + r, yi = i >> 5, xi = i & 31;
        int dy = clamp7(yi - yj), dx = clamp7(xi - xj);
        int bk1 = (dy + 7) * 15 + (dx + 7);
        int bk2 = 224 - bk1;
        float v = s[r]
            + (float)skT[(rb + r) * LPAD + bk1]
            + (float)SQ[((long)bh * NN + j) * LPAD + bk2];
        float p = __expf(v - rowm[r]) * rinv[r];
        atomicAdd(&wsF[(rb + r) * 256 + bk1], p);   // bucket scatter (ds_add_f32)
        pst[(rb + r) * 32 + sub * 16 + col] = (__bf16)p;
      }
    }
    __syncthreads();
    v16bf pa = load_frag(&pst[0], 32, lane);
    for (int nt = 0; nt < 3; ++nt) {
      const __bf16* vb = VT + ((long)bh * HD + nt * 16) * NN + jt * 32;
      v16bf bv = load_frag(vb, NN, lane);
      oacc[nt] = wmma_bf16(pa, bv, oacc[nt]);
    }
    __syncthreads();
  }
  for (int idx = lane; idx < 16 * 256; idx += 32) wsB[idx] = (__bf16)wsF[idx];
  __syncthreads();
  for (int k0 = 0; k0 < LK; k0 += 32) {
    v16bf wa = load_frag(&wsB[0] + k0, 256, lane);
    for (int nt = 0; nt < 3; ++nt) {
      const __bf16* tb = TVT + ((long)(h * HD + nt * 16)) * LK + k0;
      v16bf bv = load_frag(tb, LK, lane);
      oacc[nt] = wmma_bf16(wa, bv, oacc[nt]);
    }
  }
  for (int nt = 0; nt < 3; ++nt) {
    int d = nt * 16 + col;
    for (int r = 0; r < 8; ++r) {
      int i = i0 + rb + r;
      OUTP[((long)(b_ * NN + i)) * Cc + h * HD + d] = (__bf16)oacc[nt][r];
    }
  }
}

// ---------------- output projection: [4096,384] x [384,384] + bias ----------------
__global__ void __launch_bounds__(256) k_proj(const __bf16* OUTP, const __bf16* WPT,
                                              const float* bproj, float* out) {
  int lane = threadIdx.x & 31, w = threadIdx.x >> 5;
  int m0 = blockIdx.x * 16;
  int n0 = blockIdx.y * 128 + w * 16;
  v8f acc = {};
  for (int k0 = 0; k0 < Cc; k0 += 32) {
    v16bf a = load_frag(OUTP + (long)m0 * Cc + k0, Cc, lane);
    v16bf b = load_frag(WPT + (long)n0 * Cc + k0, Cc, lane);
    acc = wmma_bf16(a, b, acc);
  }
  int col = lane & 15, rb = (lane >> 4) << 3;
  int o = n0 + col;
  float bias = bproj[o];
  for (int r = 0; r < 8; ++r) {
    int gi = m0 + rb + r;
    int b_ = gi >> 10, ii = gi & (NN - 1);
    out[((long)(b_ * Cc + o)) * NN + ii] = acc[r] + bias;
  }
}

extern "C" void kernel_launch(void* const* d_in, const int* in_sizes, int n_in,
                              void* d_out, int out_size, void* d_ws, size_t ws_size,
                              hipStream_t stream) {
  const float* x       = (const float*)d_in[0];
  const float* w_qkv   = (const float*)d_in[1];
  const float* w_proj  = (const float*)d_in[2];
  const float* b_proj  = (const float*)d_in[3];
  const float* table_q = (const float*)d_in[4];
  const float* table_k = (const float*)d_in[5];
  const float* table_v = (const float*)d_in[6];

  char* p = (char*)d_ws;
  auto alloc = [&](size_t bytes) {
    char* r = p;
    p += (bytes + 255) & ~(size_t)255;
    return r;
  };
  __bf16* XT   = (__bf16*)alloc((size_t)B_ * NN * Cc * 2);
  __bf16* WQ   = (__bf16*)alloc((size_t)C3 * Cc * 2);
  __bf16* WPT  = (__bf16*)alloc((size_t)Cc * Cc * 2);
  __bf16* Q    = (__bf16*)alloc((size_t)B_ * HH * NN * LDH * 2);
  __bf16* K    = (__bf16*)alloc((size_t)B_ * HH * NN * LDH * 2);
  __bf16* VT   = (__bf16*)alloc((size_t)B_ * HH * HD * NN * 2);
  __bf16* TK   = (__bf16*)alloc((size_t)HH * LPAD * LDH * 2);
  __bf16* TQ   = (__bf16*)alloc((size_t)HH * LPAD * LDH * 2);
  __bf16* TVT  = (__bf16*)alloc((size_t)HH * HD * LK * 2);
  __bf16* SK   = (__bf16*)alloc((size_t)B_ * HH * NN * LPAD * 2);
  __bf16* SQ   = (__bf16*)alloc((size_t)B_ * HH * NN * LPAD * 2);
  float*  Mrow = (float*)alloc((size_t)B_ * HH * NN * 4);
  float*  Lrow = (float*)alloc((size_t)B_ * HH * NN * 4);
  __bf16* OUTP = (__bf16*)alloc((size_t)B_ * NN * Cc * 2);
  size_t total = (size_t)(p - (char*)d_ws);

  // 1. zero all workspace (provides K/N zero-padding everywhere)
  k_zero<<<1024, 256, 0, stream>>>((uint4*)d_ws, (long)((total + 15) / 16));

  // 2. conversions / transposes to bf16 operand layouts
  k_cvt_wqkv<<<(C3 * Cc + 255) / 256, 256, 0, stream>>>(w_qkv, WQ);
  k_cvt_wproj<<<(Cc * Cc + 255) / 256, 256, 0, stream>>>(w_proj, WPT);
  k_cvt_tab<<<(HH * LBK * HD + 255) / 256, 256, 0, stream>>>(table_q, table_k, table_v,
                                                             TQ, TK, TVT);
  k_cvt_x<<<(B_ * NN * Cc + 255) / 256, 256, 0, stream>>>(x, XT);

  // 3. QKV projection (WMMA)
  k_qkv<<<dim3((B_ * NN) / 16, C3 / 128), 256, 0, stream>>>(XT, WQ, Q, K, VT);

  // 4. RPE score GEMMs (WMMA)
  k_rpe<<<(B_ * HH * 64 * 15) / 8, 256, 0, stream>>>(Q, K, TK, TQ, SK, SQ);

  // 5. softmax statistics (WMMA + shuffle reductions + LDS-staged SK)
  k_pass1<<<(B_ * HH * 64) / 8, 256, 0, stream>>>(Q, K, SK, SQ, Mrow, Lrow);

  // 6. attention output + rpe_v bucket GEMM (WMMA + LDS atomics)
  k_pass2<<<B_ * HH * 64, 32, 0, stream>>>(Q, K, VT, SK, SQ, TVT, Mrow, Lrow, OUTP);

  // 7. output projection (WMMA)
  k_proj<<<dim3((B_ * NN) / 16, Cc / 128), 256, 0, stream>>>(OUTP, WPT, b_proj,
                                                             (float*)d_out);
}
